// NeighborAttention_77584289235258
// MI455X (gfx1250) — compile-verified, matrix operand order checked by reading, not run on hardware
//
#include <hip/hip_runtime.h>
#include <hip/hip_bf16.h>

// ---------------- problem constants (from reference) ----------------
#define E_EDGES   800000
#define N_EGO     50000
#define DIMF      128          // EGO_DIM == REL_DIM == HEADS*D_K
#define TEMP_SCL  0.125f       // 1/sqrt(64)
#define LN_EPS    1e-5f

// ---------------- LDS layout (bytes) ----------------
#define WQT_OFF   0            // Wq^T bf16 [128 n][128 k] (k-swizzled) -> 32768 B
#define WKT_OFF   32768        // Wk^T bf16                              -> 32768 B
#define EGO_OFF   65536        // LN(ego) bf16 [8 waves][16 rows][128]   (swizzled)
#define REL_OFF   98304        // LN(rel) bf16
#define LG_OFF    131072       // logits f32 [8][16][2]
#define BI_OFF    132096       // bias   f32 [8][16][2]
#define PRM_OFF   133120       // params: g_q(512B) b_q(512) g_k(512) b_k(512) Wb(1024)
#define SMEM_BYTES 136192

// ---------------- workspace layout (bytes) ----------------
#define WS_WBUF   0                       // float [E][2]
#define WS_SEGK   (E_EDGES * 2 * 4)       // unsigned [N][2]
#define WS_DEN    (WS_SEGK + N_EGO * 2 * 4) // float [N][2]
#define WS_WTAB   (WS_DEN + N_EGO * 2 * 4)  // bf16 swizzled WqT(32768) ++ WkT(32768)

// async global->LDS copy (CDNA5 GLOBAL_LOAD_ASYNC_TO_LDS_B128, ASYNCcnt-tracked)
#if defined(__has_builtin)
#if __has_builtin(__builtin_amdgcn_global_load_async_to_lds_b128)
#define HAVE_ASYNC_LDS 1
#endif
#if __has_builtin(__builtin_amdgcn_s_wait_asynccnt)
#define HAVE_WAIT_ASYNC 1
#endif
#endif

typedef int v4i_t __attribute__((vector_size(16)));
typedef __attribute__((address_space(1))) v4i_t* g_v4i_ptr;   // global (AS1) int4*
typedef __attribute__((address_space(3))) v4i_t* l_v4i_ptr;   // LDS    (AS3) int4*

typedef __attribute__((ext_vector_type(16))) __bf16 v16bf;
typedef __attribute__((ext_vector_type(8)))  float  v8f;

union AFrag { unsigned u[8]; v16bf v; };

__device__ __forceinline__ unsigned pack_bf2(float x0, float x1) {
  __hip_bfloat162 h2 = __float22bfloat162_rn(make_float2(x0, x1)); // v_cvt_pk_bf16_f32
  unsigned p; __builtin_memcpy(&p, &h2, 4);
  return p;
}
// monotonic float -> unsigned key (so unsigned atomicMax == float max)
__device__ __forceinline__ unsigned encKey(float f) {
  unsigned b = __float_as_uint(f);
  return (b & 0x80000000u) ? ~b : (b | 0x80000000u);
}
__device__ __forceinline__ float decKey(unsigned k) {
  unsigned b = (k & 0x80000000u) ? (k ^ 0x80000000u) : ~k;
  return __uint_as_float(b);
}
// byte offset (within a 256B row) where the bf16 pair for even column c0 lives,
// swizzled so each lane's A fragment (ISA 7.12.2 16-bit A 16x32 map) is two
// contiguous b128 reads: dword = kt*16 + hh*8 + j
__device__ __forceinline__ int swzOff(int c0) {
  int kt = c0 >> 5;
  int q  = (c0 & 31) >> 1;                 // bits: [1:0]=j01 [2]=hh [3]=j2
  int j  = ((q >> 3) << 2) | (q & 3);
  int hh = (q >> 2) & 1;
  return kt * 64 + hh * 32 + j * 4;
}

// ---------------- kernel 0: init segment max/denom ----------------
__global__ void na_init(unsigned* __restrict__ segkey, float* __restrict__ denom) {
  int i = blockIdx.x * blockDim.x + threadIdx.x;
  if (i < N_EGO * 2) { segkey[i] = 0u; denom[i] = 0.0f; }
}

// ---------------- kernel 0b: one-time bf16 swizzled weight tables ----------------
__global__ void na_prep(const float* __restrict__ Wq, const float* __restrict__ Wk,
                        char* __restrict__ wtab) {
  int p = blockIdx.x * blockDim.x + threadIdx.x;   // pair index: 128 n x 64 kpairs
  if (p >= 128 * 64) return;
  int n = p & 127, k = (p >> 7) * 2;
  int off = n * 256 + swzOff(k);
  *(unsigned*)(wtab + off)         = pack_bf2(Wq[k * DIMF + n], Wq[(k + 1) * DIMF + n]);
  *(unsigned*)(wtab + 32768 + off) = pack_bf2(Wk[k * DIMF + n], Wk[(k + 1) * DIMF + n]);
}

// ---------------- kernel 1: LN + WMMA projections + logits + seg-max ----------------
__global__ __launch_bounds__(256)
void na_main(const float* __restrict__ ego_feat, const float* __restrict__ rel_feat,
             const char* __restrict__ wtab,
             const float* __restrict__ Wb,
             const float* __restrict__ g_q, const float* __restrict__ b_q,
             const float* __restrict__ g_k, const float* __restrict__ b_k,
             const int* __restrict__ ego_idx,
             float* __restrict__ wbuf, unsigned* __restrict__ segkey) {
  extern __shared__ char smem[];
  const int tid  = threadIdx.x;
  const int wave = tid >> 5;
  const int lane = tid & 31;

  // ---- issue async DMA: params (3KB) + pre-swizzled bf16 weights (64KB) -> LDS ----
  if (tid < 192) {
    const float* srcp;
    int c = tid;
    if      (c < 32)  srcp = g_q + c * 4;
    else if (c < 64)  srcp = b_q + (c - 32) * 4;
    else if (c < 96)  srcp = g_k + (c - 64) * 4;
    else if (c < 128) srcp = b_k + (c - 96) * 4;
    else              srcp = Wb  + (c - 128) * 4;
    char* lp = smem + PRM_OFF + c * 16;
#ifdef HAVE_ASYNC_LDS
    __builtin_amdgcn_global_load_async_to_lds_b128((g_v4i_ptr)srcp, (l_v4i_ptr)lp, 0, 0);
#else
    *(float4*)lp = *(const float4*)srcp;
#endif
  }
  for (int c = tid; c < 4096; c += 256) {          // 64KB weight tables, 16 b128/thread
#ifdef HAVE_ASYNC_LDS
    __builtin_amdgcn_global_load_async_to_lds_b128(
        (g_v4i_ptr)(wtab + (size_t)c * 16), (l_v4i_ptr)(smem + WQT_OFF + c * 16), 0, 0);
#else
    *(uint4*)(smem + WQT_OFF + c * 16) = *(const uint4*)(wtab + (size_t)c * 16);
#endif
  }

  // ---- LN stats passes (overlap global-load latency with async weight DMA) ----
  const int r  = lane >> 1;           // row within wave tile (0..15)
  const int hf = lane & 1;            // which half of the 128-wide row
  const int e  = blockIdx.x * 128 + wave * 16 + r;
  const int ec = (e < E_EDGES) ? e : (E_EDGES - 1);

  // prefetch next block's ego rows (global_prefetch_b8; speculative, OOB-safe)
  __builtin_prefetch(ego_feat + (size_t)(blockIdx.x * 128 + 128) * DIMF, 0, 1);

  const float* srcE = ego_feat + ec * DIMF + hf * 64;
  const float* srcR = rel_feat + ec * DIMF + hf * 64;
  float muE, rsE, muR, rsR;
  {
    float s = 0.f, s2 = 0.f;
    #pragma unroll
    for (int j = 0; j < 64; j += 4) {
      float4 v4 = *(const float4*)(srcE + j);
      s  += v4.x + v4.y + v4.z + v4.w;
      s2 += v4.x * v4.x + v4.y * v4.y + v4.z * v4.z + v4.w * v4.w;
    }
    s  += __shfl_xor(s, 1, 32);
    s2 += __shfl_xor(s2, 1, 32);
    muE = s * (1.f / 128.f);
    rsE = rsqrtf(s2 * (1.f / 128.f) - muE * muE + LN_EPS);
  }
  {
    float s = 0.f, s2 = 0.f;
    #pragma unroll
    for (int j = 0; j < 64; j += 4) {
      float4 v4 = *(const float4*)(srcR + j);
      s  += v4.x + v4.y + v4.z + v4.w;
      s2 += v4.x * v4.x + v4.y * v4.y + v4.z * v4.z + v4.w * v4.w;
    }
    s  += __shfl_xor(s, 1, 32);
    s2 += __shfl_xor(s2, 1, 32);
    muR = s * (1.f / 128.f);
    rsR = rsqrtf(s2 * (1.f / 128.f) - muR * muR + LN_EPS);
  }

  // all async LDS data (params + weights) must have landed before use
#ifdef HAVE_ASYNC_LDS
#ifdef HAVE_WAIT_ASYNC
  __builtin_amdgcn_s_wait_asynccnt(0);
#else
  asm volatile("s_wait_asynccnt 0x0" ::: "memory");
#endif
#endif
  __syncthreads();

  const float* gqL = (const float*)(smem + PRM_OFF);          // [128]
  const float* bqL = (const float*)(smem + PRM_OFF + 512);    // [128]
  const float* gkL = (const float*)(smem + PRM_OFF + 1024);   // [128]
  const float* bkL = (const float*)(smem + PRM_OFF + 1536);   // [128]
  const float* wbL = (const float*)(smem + PRM_OFF + 2048);   // [128][2]

  // ---- normalize -> bf16 LDS tiles (swizzled); fuse raw-rel @ Wb bias ----
  #pragma unroll
  for (int j = 0; j < 64; j += 2) {
    int c0 = hf * 64 + j;
    float x0 = (srcE[j]     - muE) * rsE * gqL[c0]     + bqL[c0];
    float x1 = (srcE[j + 1] - muE) * rsE * gqL[c0 + 1] + bqL[c0 + 1];
    *(unsigned*)(smem + EGO_OFF + wave * 4096 + r * 256 + swzOff(c0)) = pack_bf2(x0, x1);
  }
  {
    float bb0 = 0.f, bb1 = 0.f;
    #pragma unroll
    for (int j = 0; j < 64; j += 2) {
      int c0 = hf * 64 + j;
      float v0 = srcR[j], v1 = srcR[j + 1];
      bb0 += v0 * wbL[c0 * 2]     + v1 * wbL[(c0 + 1) * 2];
      bb1 += v0 * wbL[c0 * 2 + 1] + v1 * wbL[(c0 + 1) * 2 + 1];
      float x0 = (v0 - muR) * rsR * gkL[c0]     + bkL[c0];
      float x1 = (v1 - muR) * rsR * gkL[c0 + 1] + bkL[c0 + 1];
      *(unsigned*)(smem + REL_OFF + wave * 4096 + r * 256 + swzOff(c0)) = pack_bf2(x0, x1);
    }
    bb0 += __shfl_xor(bb0, 1, 32);
    bb1 += __shfl_xor(bb1, 1, 32);
    if (hf == 0) {
      *(float*)(smem + BI_OFF + (wave * 16 + r) * 8)     = bb0;
      *(float*)(smem + BI_OFF + (wave * 16 + r) * 8 + 4) = bb1;
    }
  }
  __syncthreads();

  // ---- WMMA: q = LN(ego)@Wq, k = LN(rel)@Wk, logits = sum_d q*k per head ----
  const int  MN = lane & 15;          // A-row / B-col index owned by this lane
  const int  hh = lane >> 4;          // half-wave selector in K mapping
  const char* egoBase = smem + EGO_OFF + wave * 4096 + MN * 256 + (hh << 5);
  const char* relBase = smem + REL_OFF + wave * 4096 + MN * 256 + (hh << 5);

  // hoist A fragments (reused by all 8 column tiles): 2x b128 per ktile per matrix
  AFrag aE[4], aR[4];
  #pragma unroll
  for (int kt = 0; kt < 4; ++kt) {
    *(uint4*)&aE[kt].u[0] = *(const uint4*)(egoBase + kt * 64);
    *(uint4*)&aE[kt].u[4] = *(const uint4*)(egoBase + kt * 64 + 16);
    *(uint4*)&aR[kt].u[0] = *(const uint4*)(relBase + kt * 64);
    *(uint4*)&aR[kt].u[4] = *(const uint4*)(relBase + kt * 64 + 16);
  }

  v8f acc0 = {};  // head 0 partial q·k products, C-matrix layout
  v8f acc1 = {};  // head 1
  #pragma unroll
  for (int t = 0; t < 8; ++t) {                 // 8 column tiles of 16 (4 per head)
    v8f q = {}, kv = {};
    const char* bqBase = smem + WQT_OFF + (t * 16 + MN) * 256 + (hh << 5);
    const char* bkBase = smem + WKT_OFF + (t * 16 + MN) * 256 + (hh << 5);
    #pragma unroll
    for (int kt = 0; kt < 4; ++kt) {            // K = 128 in chunks of 32
      AFrag b, b2;
      *(uint4*)&b.u[0]  = *(const uint4*)(bqBase + kt * 64);
      *(uint4*)&b.u[4]  = *(const uint4*)(bqBase + kt * 64 + 16);
      *(uint4*)&b2.u[0] = *(const uint4*)(bkBase + kt * 64);
      *(uint4*)&b2.u[4] = *(const uint4*)(bkBase + kt * 64 + 16);
      q  = __builtin_amdgcn_wmma_f32_16x16x32_bf16(false, aE[kt].v, false, b.v,  (short)0, q,  false, false);
      kv = __builtin_amdgcn_wmma_f32_16x16x32_bf16(false, aR[kt].v, false, b2.v, (short)0, kv, false, false);
    }
    #pragma unroll
    for (int rr = 0; rr < 8; ++rr) {
      float p = q[rr] * kv[rr];
      if (t < 4) acc0[rr] += p; else acc1[rr] += p;
    }
  }

  // ---- reduce over N within each half-wave (C layout: lanes 0-15 = M=rr, 16-31 = M=rr+8) ----
  #pragma unroll
  for (int rr = 0; rr < 8; ++rr) {
    float a0 = acc0[rr], a1 = acc1[rr];
    #pragma unroll
    for (int m = 1; m < 16; m <<= 1) {
      a0 += __shfl_xor(a0, m, 32);
      a1 += __shfl_xor(a1, m, 32);
    }
    if (lane == 0) {
      *(float*)(smem + LG_OFF + (wave * 16 + rr) * 8)     = a0;
      *(float*)(smem + LG_OFF + (wave * 16 + rr) * 8 + 4) = a1;
    }
    if (lane == 16) {
      *(float*)(smem + LG_OFF + (wave * 16 + rr + 8) * 8)     = a0;
      *(float*)(smem + LG_OFF + (wave * 16 + rr + 8) * 8 + 4) = a1;
    }
  }
  __syncthreads();

  // ---- finalize logits, write scratch, segmented atomic max ----
  if (lane < 16) {
    int e2 = blockIdx.x * 128 + wave * 16 + lane;
    if (e2 < E_EDGES) {
      float bi0 = *(float*)(smem + BI_OFF + (wave * 16 + lane) * 8);
      float bi1 = *(float*)(smem + BI_OFF + (wave * 16 + lane) * 8 + 4);
      float l0  = *(float*)(smem + LG_OFF + (wave * 16 + lane) * 8)     * TEMP_SCL + bi0;
      float l1  = *(float*)(smem + LG_OFF + (wave * 16 + lane) * 8 + 4) * TEMP_SCL + bi1;
      wbuf[e2 * 2 + 0] = l0;
      wbuf[e2 * 2 + 1] = l1;
      int ego = ego_idx[e2];
      atomicMax(&segkey[ego * 2 + 0], encKey(l0));
      atomicMax(&segkey[ego * 2 + 1], encKey(l1));
    }
  }
}

// ---------------- kernel 2: w = exp(l - segmax); denom += w ----------------
__global__ void na_exp(float* __restrict__ wbuf, const unsigned* __restrict__ segkey,
                       float* __restrict__ denom, const int* __restrict__ ego_idx) {
  int e = blockIdx.x * blockDim.x + threadIdx.x;
  if (e >= E_EDGES) return;
  int ego = ego_idx[e];
  #pragma unroll
  for (int h = 0; h < 2; ++h) {
    float l = wbuf[e * 2 + h];
    float w = __expf(l - decKey(segkey[ego * 2 + h]));
    wbuf[e * 2 + h] = w;
    atomicAdd(&denom[ego * 2 + h], w);     // global_atomic_add_f32
  }
}

// ---------------- kernel 3: alpha = mean_h w/(denom+1e-6) ----------------
__global__ void na_norm(const float* __restrict__ wbuf, const float* __restrict__ denom,
                        const int* __restrict__ ego_idx, float* __restrict__ out) {
  int e = blockIdx.x * blockDim.x + threadIdx.x;
  if (e >= E_EDGES) return;
  int ego = ego_idx[e];
  float a0 = wbuf[e * 2 + 0] / (denom[ego * 2 + 0] + 1e-6f);
  float a1 = wbuf[e * 2 + 1] / (denom[ego * 2 + 1] + 1e-6f);
  out[e] = 0.5f * (a0 + a1);
}

extern "C" void kernel_launch(void* const* d_in, const int* in_sizes, int n_in,
                              void* d_out, int out_size, void* d_ws, size_t ws_size,
                              hipStream_t stream) {
  const float* ego_feat = (const float*)d_in[0];
  const float* rel_feat = (const float*)d_in[1];
  const float* Wq       = (const float*)d_in[2];
  const float* Wk       = (const float*)d_in[3];
  const float* Wb       = (const float*)d_in[4];
  const float* g_q      = (const float*)d_in[5];
  const float* b_q      = (const float*)d_in[6];
  const float* g_k      = (const float*)d_in[7];
  const float* b_k      = (const float*)d_in[8];
  const int*   ego_idx  = (const int*)d_in[9];
  float*       out      = (float*)d_out;

  float*    wbuf   = (float*)((char*)d_ws + WS_WBUF);
  unsigned* segkey = (unsigned*)((char*)d_ws + WS_SEGK);
  float*    denom  = (float*)((char*)d_ws + WS_DEN);
  char*     wtab   = (char*)d_ws + WS_WTAB;

  na_init<<<(N_EGO * 2 + 255) / 256, 256, 0, stream>>>(segkey, denom);
  na_prep<<<(128 * 64 + 255) / 256, 256, 0, stream>>>(Wq, Wk, wtab);
  na_main<<<(E_EDGES + 127) / 128, 256, SMEM_BYTES, stream>>>(
      ego_feat, rel_feat, wtab, Wb, g_q, b_q, g_k, b_k, ego_idx, wbuf, segkey);
  na_exp<<<(E_EDGES + 255) / 256, 256, 0, stream>>>(wbuf, segkey, denom, ego_idx);
  na_norm<<<(E_EDGES + 255) / 256, 256, 0, stream>>>(wbuf, denom, ego_idx, out);
}
